// HGAT_LSTM_19868518711932
// MI455X (gfx1250) — matrix operand order repaired and checked by reading, not compiled, not run on hardware
//
#include <hip/hip_runtime.h>

// ---------------- model constants ----------------
#define BB   32
#define TT   12
#define NN   128
#define DD   16
#define LLAY 2
#define GG   3
#define NHH  4
#define WDD  8
#define EE   2048
#define CNN  64
#define BT   (BB*TT)      // 384
#define HH   (NN*DD)      // 2048
#define H4   (4*HH)       // 8192
#define KCH  256          // LSTM K-chunk staged in LDS

typedef __attribute__((ext_vector_type(16))) _Float16 v16h;
typedef __attribute__((ext_vector_type(8)))  _Float16 v8h;
typedef __attribute__((ext_vector_type(8)))  float    v8f;
typedef __attribute__((ext_vector_type(4)))  int      v4i;

#define CAT16(lo, hi) __builtin_shufflevector(lo, hi, 0,1,2,3,4,5,6,7,8,9,10,11,12,13,14,15)

// ---- CDNA5 async global->LDS path (probe via __has_builtin; safe fallback) ----
// Diagnosed signature: (v4i addrspace(1)* src, v4i addrspace(3)* dst, imm off, imm cpol)
#if __has_builtin(__builtin_amdgcn_global_load_async_to_lds_b128) && \
    __has_builtin(__builtin_amdgcn_s_wait_asynccnt)
#define HAVE_ASYNC_LDS 1
typedef __attribute__((address_space(1))) v4i as1_v4i;
typedef __attribute__((address_space(3))) v4i as3_v4i;
#else
#define HAVE_ASYNC_LDS 0
#endif

// 16-lane xor butterfly reduction via ds_swizzle (group-of-32 mode, and=0x1f)
__device__ __forceinline__ float xor16_reduce(float v) {
  v += __int_as_float(__builtin_amdgcn_ds_swizzle(__float_as_int(v), (1 << 10) | 0x1f));
  v += __int_as_float(__builtin_amdgcn_ds_swizzle(__float_as_int(v), (2 << 10) | 0x1f));
  v += __int_as_float(__builtin_amdgcn_ds_swizzle(__float_as_int(v), (4 << 10) | 0x1f));
  v += __int_as_float(__builtin_amdgcn_ds_swizzle(__float_as_int(v), (8 << 10) | 0x1f));
  return v;
}

// =====================================================================
// K0: stable CSR build per graph (dst -> edge list), deterministic.
// =====================================================================
__global__ void k_build_csr(const int* __restrict__ dst_all,
                            int* __restrict__ rowptr_all,
                            int* __restrict__ eidx_all) {
  const int g = blockIdx.x;
  const int* dst = dst_all + g * EE;
  int* rowptr = rowptr_all + g * (NN + 1);
  int* eidx   = eidx_all   + g * EE;
  __shared__ int cnt[NN + 1];
  __shared__ int roff[NN + 1];
  for (int i = threadIdx.x; i <= NN; i += blockDim.x) cnt[i] = 0;
  __syncthreads();
  for (int e = threadIdx.x; e < EE; e += blockDim.x) atomicAdd(&cnt[dst[e]], 1);
  __syncthreads();
  if (threadIdx.x == 0) {
    int acc = 0;
    for (int n = 0; n <= NN; ++n) { roff[n] = acc; acc += cnt[n]; }
  }
  __syncthreads();
  for (int i = threadIdx.x; i <= NN; i += blockDim.x) rowptr[i] = roff[i];
  for (int e = threadIdx.x; e < EE; e += blockDim.x) {
    int d = dst[e];
    int rank = 0;                       // stable: count prior edges with same dst
    for (int j = 0; j < e; ++j) rank += (dst[j] == d) ? 1 : 0;
    eidx[roff[d] + rank] = e;
  }
}

// =====================================================================
// K_ee: ee[g,h,e] = dot(ew[g,e,:WD], w_e[g,h,:WD])
// =====================================================================
__global__ void k_ee(const float* __restrict__ ew,
                     const float* __restrict__ we,
                     float* __restrict__ eebuf) {
  int i = blockIdx.x * blockDim.x + threadIdx.x;
  if (i >= GG * NHH * EE) return;
  int e = i % EE, h = (i / EE) % NHH, g = i / (EE * NHH);
  float acc = 0.f;
  #pragma unroll
  for (int w = 0; w < WDD; ++w)
    acc += ew[(g * EE + e) * WDD + w] * we[(g * NHH + h) * WDD + w];
  eebuf[i] = acc;
}

// =====================================================================
// K_prevec: fold attention vectors through W:  wsv = W@a_s, wdv = W@a_d
// so es/ed come straight out of a WMMA (es = x·(W@a_s)).
// =====================================================================
__global__ void k_prevec(const float* __restrict__ W,    // [G,NH,D,D]
                         const float* __restrict__ a_s,  // [G,NH,D]
                         const float* __restrict__ a_d,  // [G,NH,D]
                         float* __restrict__ wsv,        // [G*NH,D]
                         float* __restrict__ wdv) {
  int i = blockIdx.x * blockDim.x + threadIdx.x;
  if (i >= GG * NHH * DD) return;
  const int d = i % DD, gh = i / DD;
  const float* wr = W + (size_t)gh * DD * DD + (size_t)d * DD;
  float s = 0.f, t = 0.f;
  #pragma unroll
  for (int o = 0; o < DD; ++o) {
    s += wr[o] * a_s[gh * DD + o];
    t += wr[o] * a_d[gh * DD + o];
  }
  wsv[i] = s;
  wdv[i] = t;
}

// =====================================================================
// K1: z = x @ W via WMMA; second WMMA emits es/ed (cols 0/1). No shuffles.
// grid: (BT*N/16/8, G*NH), block 256 (8 waves).
// =====================================================================
__global__ __launch_bounds__(256)
void k_gat_z(const float* __restrict__ hin,     // [BT,N,D]
             const float* __restrict__ W,       // [G,NH,D,D]
             const float* __restrict__ wsv,     // [G*NH,D]
             const float* __restrict__ wdv,     // [G*NH,D]
             _Float16* __restrict__ zbuf,       // [G,NH,BT,N,D] f16
             float* __restrict__ esbuf,         // [G,NH,BT,N]
             float* __restrict__ edbuf) {
  const int lane = threadIdx.x & 31;
  const int wave = threadIdx.x >> 5;
  const int tile = blockIdx.x * 8 + wave;
  const int gh   = blockIdx.y;
  const int row0 = tile * 16;
  const int hf   = lane >> 4;
  const int lm   = lane & 15;

  // A: 16x32 f16 (K 0..15 valid, 16..31 zero-pad)
  const float* xr = hin + (size_t)(row0 + lm) * DD;
  v16h A;
  #pragma unroll
  for (int i = 0; i < 16; ++i) A[i] = hf ? (_Float16)0.f : (_Float16)xr[i];

  // B1: W (lane = out col, K contiguous rows; lanes>=16 are K 16..31 pad = 0)
  const float* wb = W + (size_t)gh * DD * DD;
  v16h B1;
  #pragma unroll
  for (int k = 0; k < 16; ++k)
    B1[k] = hf ? (_Float16)0.f : (_Float16)wb[k * DD + lm];

  v8f C = {};
  C = __builtin_amdgcn_wmma_f32_16x16x32_f16(false, A, false, B1, (short)0, C, false, false);

  // B2: col0 = W@a_s, col1 = W@a_d -> C2 col0 = es, col1 = ed
  const float* vs = wsv + gh * DD;
  const float* vd = wdv + gh * DD;
  v16h B2;
  #pragma unroll
  for (int k = 0; k < 16; ++k) {
    float bv = 0.f;
    if (hf == 0 && lm == 0) bv = vs[k];
    if (hf == 0 && lm == 1) bv = vd[k];
    B2[k] = (_Float16)bv;
  }
  v8f C2 = {};
  C2 = __builtin_amdgcn_wmma_f32_16x16x32_f16(false, A, false, B2, (short)0, C2, false, false);

  const int bt = row0 / NN;
  const int n0 = row0 % NN;
  const size_t zbase = ((size_t)gh * BT + bt) * NN * DD;
  #pragma unroll
  for (int r = 0; r < 8; ++r) {
    zbuf[zbase + (size_t)(n0 + r + hf * 8) * DD + lm] = (_Float16)C[r];
    const int ridx = (gh * BT + bt) * NN + n0 + r + hf * 8;
    if (lm == 0) esbuf[ridx] = C2[r];
    if (lm == 1) edbuf[ridx] = C2[r];
  }
}

// =====================================================================
// K2: per-destination edge softmax + aggregation, mean over heads.
// =====================================================================
__global__ void k_gat_aggr(const float* __restrict__ esbuf,
                           const float* __restrict__ edbuf,
                           const float* __restrict__ eebuf,
                           const _Float16* __restrict__ zbuf,
                           const int* __restrict__ src_all,
                           const int* __restrict__ rowptr_all,
                           const int* __restrict__ eidx_all,
                           float* __restrict__ gat_out) {
  int i = blockIdx.x * blockDim.x + threadIdx.x;
  if (i >= GG * NN * BT) return;
  const int bt = i % BT;
  const int n  = (i / BT) % NN;
  const int g  = i / (BT * NN);
  const int* rp   = rowptr_all + g * (NN + 1);
  const int* ei   = eidx_all   + g * EE;
  const int* srcv = src_all    + g * EE;
  const int e0 = rp[n], e1 = rp[n + 1];

  float acc[DD];
  #pragma unroll
  for (int d = 0; d < DD; ++d) acc[d] = 0.f;

  for (int h = 0; h < NHH; ++h) {
    const int ghb = g * NHH + h;
    const float* es = esbuf + (size_t)ghb * BT * NN + (size_t)bt * NN;
    const float* ee = eebuf + (size_t)ghb * EE;
    const float edv = edbuf[(size_t)ghb * BT * NN + (size_t)bt * NN + n];
    float mx = -1e30f;
    for (int p = e0; p < e1; ++p) {
      const int e = ei[p];
      float v = es[srcv[e]] + edv + ee[e];
      v = (v > 0.f) ? v : 0.2f * v;
      mx = fmaxf(mx, v);
    }
    if (e1 == e0) mx = 0.f;
    float den = 0.f;
    float num[DD];
    #pragma unroll
    for (int d = 0; d < DD; ++d) num[d] = 0.f;
    const _Float16* zb = zbuf + ((size_t)ghb * BT + bt) * NN * DD;
    for (int p = e0; p < e1; ++p) {
      const int e = ei[p];
      const int s = srcv[e];
      float v = es[s] + edv + ee[e];
      v = (v > 0.f) ? v : 0.2f * v;
      const float ex = __expf(v - mx);
      den += ex;
      const _Float16* zr = zb + (size_t)s * DD;
      #pragma unroll
      for (int d = 0; d < DD; ++d) num[d] += ex * (float)zr[d];
    }
    const float inv = 1.f / (den + 1e-9f);
    #pragma unroll
    for (int d = 0; d < DD; ++d) acc[d] += num[d] * inv;
  }
  float* o = gat_out + ((size_t)g * BT + bt) * NN * DD + (size_t)n * DD;
  #pragma unroll
  for (int d = 0; d < DD; ++d) o[d] = acc[d] * (1.f / NHH);
}

// =====================================================================
// K3: LinearAttention fusion; WMMA for gat@W1^T; ds_swizzle reductions.
// =====================================================================
__global__ __launch_bounds__(256)
void k_att_fuse(const float* __restrict__ gat_out,  // [G,BT,N,D]
                const float* __restrict__ W1,       // [G,128,D]
                const float* __restrict__ b1,       // [G,128]
                const float* __restrict__ W2,       // [G,128]
                float* __restrict__ hout,           // [BT,N,D]
                int node_lo, int nodes) {
  const int lane = threadIdx.x & 31;
  const int wave = threadIdx.x >> 5;
  const int tpb  = nodes / 16;
  const int tile = blockIdx.x * 8 + wave;
  const int bt   = tile / tpb;
  const int n0   = node_lo + (tile % tpb) * 16;
  const int hf   = lane >> 4;
  const int lm   = lane & 15;
  __shared__ float lg[8][GG][16];

  for (int g = 0; g < GG; ++g) {
    const float* arow = gat_out + ((size_t)g * BT + bt) * NN * DD
                        + (size_t)(n0 + lm) * DD;
    v16h A;
    #pragma unroll
    for (int i = 0; i < 16; ++i) A[i] = hf ? (_Float16)0.f : (_Float16)arow[i];

    float lacc[8];
    #pragma unroll
    for (int r = 0; r < 8; ++r) lacc[r] = 0.f;

    #pragma unroll
    for (int jt = 0; jt < 8; ++jt) {
      const float* w1 = W1 + ((size_t)g * 128 + jt * 16 + lm) * DD;
      v16h Bm;
      #pragma unroll
      for (int k = 0; k < 16; ++k)
        Bm[k] = hf ? (_Float16)0.f : (_Float16)w1[k];
      v8f C = {};
      C = __builtin_amdgcn_wmma_f32_16x16x32_f16(false, A, false, Bm, (short)0, C,
                                                 false, false);
      const float bj = b1[g * 128 + jt * 16 + lm];
      const float w2 = W2[g * 128 + jt * 16 + lm];
      #pragma unroll
      for (int r = 0; r < 8; ++r) lacc[r] += tanhf(C[r] + bj) * w2;
    }
    #pragma unroll
    for (int r = 0; r < 8; ++r) {
      const float s = xor16_reduce(lacc[r]);
      if (lm == 0) lg[wave][g][r + hf * 8] = s;
    }
  }
  __syncthreads();
  if (lane < 16) {
    const int n = n0 + lane;
    const float l0 = lg[wave][0][lane], l1 = lg[wave][1][lane], l2 = lg[wave][2][lane];
    const float mx = fmaxf(l0, fmaxf(l1, l2));
    const float x0 = __expf(l0 - mx), x1 = __expf(l1 - mx), x2 = __expf(l2 - mx);
    const float inv = 1.f / (x0 + x1 + x2);
    const float b0 = x0 * inv, b1v = x1 * inv, b2 = x2 * inv;
    const float* g0 = gat_out + ((size_t)0 * BT + bt) * NN * DD + (size_t)n * DD;
    const float* g1 = gat_out + ((size_t)1 * BT + bt) * NN * DD + (size_t)n * DD;
    const float* g2 = gat_out + ((size_t)2 * BT + bt) * NN * DD + (size_t)n * DD;
    float* o = hout + ((size_t)bt * NN + n) * DD;
    #pragma unroll
    for (int d = 0; d < DD; ++d) o[d] = b0 * g0[d] + b1v * g1[d] + b2 * g2[d];
  }
}

// =====================================================================
// K4/K5/zero: packing + weight conversion
// =====================================================================
__global__ void k_pack_xs(const float* __restrict__ hin, _Float16* __restrict__ xs) {
  int i = blockIdx.x * blockDim.x + threadIdx.x;
  if (i >= TT * BB * HH) return;
  const int j = i % HH, b = (i / HH) % BB, t = i / (HH * BB);
  xs[i] = (_Float16)hin[((size_t)b * TT + t) * HH + j];
}

__global__ void k_w2h(const float* __restrict__ a, const float* __restrict__ b,
                      _Float16* __restrict__ oa, _Float16* __restrict__ ob, int n) {
  int i = blockIdx.x * blockDim.x + threadIdx.x;
  if (i < n) { oa[i] = (_Float16)a[i]; ob[i] = (_Float16)b[i]; }
}

__global__ void k_zero_state(_Float16* __restrict__ h16, float* __restrict__ c) {
  int i = blockIdx.x * blockDim.x + threadIdx.x;
  if (i < BB * HH) { h16[i] = (_Float16)0.f; c[i] = 0.f; }
}

// =====================================================================
// K6: LSTM gate GEMM.  A (x_t, h_prev) staged into LDS with async
// global->LDS (CDNA5 path), shared by all 8 waves of the block.
// Each wave owns 4 N-tiles (4 f32 accumulators, 8 WMMAs per k-step).
// Weights: 64MB f16 total -> L2-resident on MI455X (192MB L2).
// =====================================================================
__global__ __launch_bounds__(256)
void k_lstm_gemm(const _Float16* __restrict__ xt,      // [32,2048]
                 const _Float16* __restrict__ hprev,   // [32,2048]
                 const _Float16* __restrict__ Wih16,   // [8192,2048]
                 const _Float16* __restrict__ Whh16,   // [8192,2048]
                 float* __restrict__ gates) {          // [32,8192]
  __shared__ _Float16 Ax[16][KCH];
  __shared__ _Float16 Ah[16][KCH];
  const int lane = threadIdx.x & 31;
  const int wave = threadIdx.x >> 5;
  const int mt   = blockIdx.y;                        // 0..1
  const int ntb  = (blockIdx.x * 8 + wave) * 4;       // 4 n-tiles per wave
  const int hf   = lane >> 4;
  const int lm   = lane & 15;
  const int tid  = threadIdx.x;

  v8f Cacc[4];
  #pragma unroll
  for (int j = 0; j < 4; ++j) Cacc[j] = (v8f){};

  for (int kb = 0; kb < HH; kb += KCH) {
    __syncthreads();                                  // LDS reuse fence
    #pragma unroll
    for (int c = 0; c < 2; ++c) {
      const int chunk = tid + c * 256;                // 512 x 16B per matrix
      const int row   = chunk >> 5;
      const int ko    = (chunk & 31) * 8;
      const size_t goff = (size_t)(mt * 16 + row) * HH + kb + ko;
#if HAVE_ASYNC_LDS
      __builtin_amdgcn_global_load_async_to_lds_b128(
          (as1_v4i*)(xt + goff), (as3_v4i*)&Ax[row][ko], 0, 0);
      __builtin_amdgcn_global_load_async_to_lds_b128(
          (as1_v4i*)(hprev + goff), (as3_v4i*)&Ah[row][ko], 0, 0);
#else
      *(v8h*)&Ax[row][ko] = *(const v8h*)(xt + goff);
      *(v8h*)&Ah[row][ko] = *(const v8h*)(hprev + goff);
#endif
    }
#if HAVE_ASYNC_LDS
    __builtin_amdgcn_s_wait_asynccnt(0);
#endif
    __syncthreads();

    for (int k0 = 0; k0 < KCH; k0 += 32) {
      const int kg = kb + k0;                          // global K base
      {
        v8h lo = *(const v8h*)&Ax[lm][k0 + hf * 8];    // ds_load_b128
        v8h hi = *(const v8h*)&Ax[lm][k0 + 16 + hf * 8];
        v16h Af = CAT16(lo, hi);
        #pragma unroll
        for (int j = 0; j < 4; ++j) {
          const _Float16* bp = Wih16 + (size_t)((ntb + j) * 16 + lm) * HH + hf * 8 + kg;
          __builtin_prefetch(bp + 32, 0, 1);           // global_prefetch_b8
          v8h b0 = *(const v8h*)bp;
          v8h b1 = *(const v8h*)(bp + 16);
          v16h Bf = CAT16(b0, b1);
          Cacc[j] = __builtin_amdgcn_wmma_f32_16x16x32_f16(false, Af, false, Bf,
                                                           (short)0, Cacc[j], false, false);
        }
      }
      {
        v8h lo = *(const v8h*)&Ah[lm][k0 + hf * 8];
        v8h hi = *(const v8h*)&Ah[lm][k0 + 16 + hf * 8];
        v16h Af = CAT16(lo, hi);
        #pragma unroll
        for (int j = 0; j < 4; ++j) {
          const _Float16* bp = Whh16 + (size_t)((ntb + j) * 16 + lm) * HH + hf * 8 + kg;
          __builtin_prefetch(bp + 32, 0, 1);
          v8h b0 = *(const v8h*)bp;
          v8h b1 = *(const v8h*)(bp + 16);
          v16h Bf = CAT16(b0, b1);
          Cacc[j] = __builtin_amdgcn_wmma_f32_16x16x32_f16(false, Af, false, Bf,
                                                           (short)0, Cacc[j], false, false);
        }
      }
    }
  }
  #pragma unroll
  for (int r = 0; r < 8; ++r) {
    const size_t rowoff = (size_t)(mt * 16 + r + hf * 8) * H4 + lm;
    #pragma unroll
    for (int j = 0; j < 4; ++j)
      gates[rowoff + (size_t)(ntb + j) * 16] = Cacc[j][r];
  }
}

// =====================================================================
// K7: LSTM elementwise update (bias folded; torch gate order i,f,g,o)
// =====================================================================
__global__ void k_lstm_update(const float* __restrict__ gates,
                              const float* __restrict__ bih,
                              const float* __restrict__ bhh,
                              float* __restrict__ c,
                              _Float16* __restrict__ h16,
                              float* __restrict__ outp) {
  int i = blockIdx.x * blockDim.x + threadIdx.x;
  if (i >= BB * HH) return;
  const int b = i / HH, j = i % HH;
  const float* gr = gates + (size_t)b * H4;
  const float gi = gr[j]          + bih[j]          + bhh[j];
  const float gf = gr[HH + j]     + bih[HH + j]     + bhh[HH + j];
  const float gg = gr[2 * HH + j] + bih[2 * HH + j] + bhh[2 * HH + j];
  const float go = gr[3 * HH + j] + bih[3 * HH + j] + bhh[3 * HH + j];
  const float si = 1.f / (1.f + __expf(-gi));
  const float sf = 1.f / (1.f + __expf(-gf));
  const float so = 1.f / (1.f + __expf(-go));
  const float cn = sf * c[i] + si * tanhf(gg);
  const float hn = so * tanhf(cn);
  c[i] = cn;
  h16[i] = (_Float16)hn;
  if (outp) outp[i] = hn;
}

// =====================================================================
// host-side orchestration
// =====================================================================
extern "C" void kernel_launch(void* const* d_in, const int* in_sizes, int n_in,
                              void* d_out, int out_size, void* d_ws, size_t ws_size,
                              hipStream_t stream) {
  const float* x        = (const float*)d_in[0];
  const int*   bike_src = (const int*)d_in[1];
  const int*   bike_dst = (const int*)d_in[2];
  const float* bike_ew  = (const float*)d_in[3];
  const int*   taxi_src = (const int*)d_in[4];
  const int*   taxi_dst = (const int*)d_in[5];
  const float* taxi_ew  = (const float*)d_in[6];
  const float* gat_W    = (const float*)d_in[7];
  const float* gat_as   = (const float*)d_in[8];
  const float* gat_ad   = (const float*)d_in[9];
  const float* gat_we   = (const float*)d_in[10];
  const float* att_W1   = (const float*)d_in[11];
  const float* att_b1   = (const float*)d_in[12];
  const float* att_W2   = (const float*)d_in[13];
  const float* Wih      = (const float*)d_in[14];
  const float* Whh      = (const float*)d_in[15];
  const float* bih      = (const float*)d_in[16];
  const float* bhh      = (const float*)d_in[17];
  float* out = (float*)d_out;

  // -------- workspace carve-up --------
  char* w = (char*)d_ws;
  auto carve = [&](size_t bytes) {
    char* p = w;
    w += (bytes + 255) & ~(size_t)255;
    return p;
  };
  _Float16* zbuf   = (_Float16*)carve((size_t)GG * NHH * BT * NN * DD * 2);  // 18.9MB
  float*    esbuf  = (float*)carve((size_t)GG * NHH * BT * NN * 4);          // 2.4MB
  float*    edbuf  = (float*)carve((size_t)GG * NHH * BT * NN * 4);
  float*    eebuf  = (float*)carve((size_t)GG * NHH * EE * 4);
  float*    gat_o  = (float*)carve((size_t)GG * BT * NN * DD * 4);           // 9.4MB
  float*    hA     = (float*)carve((size_t)BT * NN * DD * 4);                // 3.1MB
  float*    hB     = (float*)carve((size_t)BT * NN * DD * 4);
  int*      rowptr = (int*)carve((size_t)2 * GG * (NN + 1) * 4);
  int*      eidx   = (int*)carve((size_t)2 * GG * EE * 4);
  float*    wsv    = (float*)carve((size_t)GG * NHH * DD * 4);
  float*    wdv    = (float*)carve((size_t)GG * NHH * DD * 4);
  _Float16* xs16   = (_Float16*)carve((size_t)TT * BB * HH * 2);             // 1.6MB
  _Float16* Wih16  = (_Float16*)carve((size_t)H4 * HH * 2);                  // 33.5MB
  _Float16* Whh16  = (_Float16*)carve((size_t)H4 * HH * 2);                  // 33.5MB
  float*    gates  = (float*)carve((size_t)BB * H4 * 4);                     // 1MB
  _Float16* h16    = (_Float16*)carve((size_t)BB * HH * 2);
  float*    cbuf   = (float*)carve((size_t)BB * HH * 4);

  // -------- CSR once (edges identical across layers) --------
  k_build_csr<<<GG, 256, 0, stream>>>(bike_dst, rowptr, eidx);
  k_build_csr<<<GG, 256, 0, stream>>>(taxi_dst, rowptr + GG * (NN + 1), eidx + GG * EE);

  // -------- HGAT layers --------
  const float* hcur = x;                 // input [B,T,N,D] == [BT,N,D] row-major
  for (int l = 0; l < LLAY; ++l) {
    float* hnext = (l == 0) ? hA : hB;
    for (int mode = 0; mode < 2; ++mode) {
      const int pm = l * 2 + mode;
      const float* Wl  = gat_W  + (size_t)pm * GG * NHH * DD * DD;
      const float* asl = gat_as + (size_t)pm * GG * NHH * DD;
      const float* adl = gat_ad + (size_t)pm * GG * NHH * DD;
      const float* wel = gat_we + (size_t)pm * GG * NHH * WDD;
      const float* W1l = att_W1 + (size_t)pm * GG * 128 * DD;
      const float* b1l = att_b1 + (size_t)pm * GG * 128;
      const float* W2l = att_W2 + (size_t)pm * GG * 128;
      const float* ewm = mode ? taxi_ew  : bike_ew;
      const int*  srcm = mode ? taxi_src : bike_src;
      const int*  rpm  = rowptr + mode * GG * (NN + 1);
      const int*  eim  = eidx   + mode * GG * EE;

      k_ee<<<(GG * NHH * EE + 255) / 256, 256, 0, stream>>>(ewm, wel, eebuf);
      k_prevec<<<1, GG * NHH * DD, 0, stream>>>(Wl, asl, adl, wsv, wdv);

      dim3 gz((BT * NN / 16) / 8, GG * NHH);          // 384 x 12 blocks
      k_gat_z<<<gz, 256, 0, stream>>>(hcur, Wl, wsv, wdv, zbuf, esbuf, edbuf);

      k_gat_aggr<<<(GG * NN * BT + 255) / 256, 256, 0, stream>>>(
          esbuf, edbuf, eebuf, zbuf, srcm, rpm, eim, gat_o);

      const int nlo  = mode ? CNN : 0;
      const int ncnt = mode ? (NN - CNN) : CNN;       // 64 each
      const int tiles = BT * (ncnt / 16);             // 1536
      k_att_fuse<<<tiles / 8, 256, 0, stream>>>(gat_o, W1l, b1l, W2l, hnext, nlo, ncnt);
    }
    hcur = hnext;
  }

  // -------- LSTM --------
  k_pack_xs<<<(TT * BB * HH + 255) / 256, 256, 0, stream>>>(hcur, xs16);
  k_w2h<<<((int)((size_t)H4 * HH) + 255) / 256, 256, 0, stream>>>(
      Wih, Whh, Wih16, Whh16, (int)((size_t)H4 * HH));
  k_zero_state<<<(BB * HH + 255) / 256, 256, 0, stream>>>(h16, cbuf);

  for (int t = 0; t < TT; ++t) {
    dim3 gl(16, 2);                                   // 32 blocks, 8 waves each
    k_lstm_gemm<<<gl, 256, 0, stream>>>(xs16 + (size_t)t * BB * HH, h16,
                                        Wih16, Whh16, gates);
    k_lstm_update<<<(BB * HH + 255) / 256, 256, 0, stream>>>(
        gates, bih, bhh, cbuf, h16, (t == TT - 1) ? out : nullptr);
  }
}
